// GlobalSparseTransformer_35244501631303
// MI455X (gfx1250) — matrix operand order repaired and checked by reading, not compiled, not run on hardware
//
#include <hip/hip_runtime.h>

// ---------------------------------------------------------------------------
// GlobalSparseTransformer for MI455X (gfx1250, wave32, WMMA bf16 -> f32)
// Double-buffered async global->LDS staging (ASYNCcnt), flash attention.
// ---------------------------------------------------------------------------
typedef __bf16 bf16_t;
typedef __attribute__((ext_vector_type(16))) __bf16 v16bf;
typedef __attribute__((ext_vector_type(8)))  __bf16 v8bf;
typedef __attribute__((ext_vector_type(8)))  float  v8f;

#define BATCH 8
#define CIN   384
#define HS    48
#define NTOK  2304      // 48*48
#define OCH   1152
#define HEADS 6
#define HDIM  64
#define QT8   (NTOK / 16 / 8)   // 18 query-tile groups per (b,h)

// Let the compiler pick the native f32->bf16 conversion (v_cvt_*_bf16_f32 on
// CDNA5); it can also pair adjacent converts into packed form.
__device__ __forceinline__ bf16_t f2bf(float f) { return (bf16_t)f; }

__device__ __forceinline__ v8f vzero8() {
  v8f r;
#pragma unroll
  for (int i = 0; i < 8; ++i) r[i] = 0.0f;
  return r;
}

__device__ __forceinline__ v16bf cat8(v8bf lo, v8bf hi) {
  v16bf r;
#pragma unroll
  for (int i = 0; i < 8; ++i) { r[i] = lo[i]; r[i + 8] = hi[i]; }
  return r;
}

__device__ __forceinline__ float rmax16(float v) {
#pragma unroll
  for (int m = 1; m < 16; m <<= 1) v = fmaxf(v, __shfl_xor(v, m));
  return v;
}
__device__ __forceinline__ float rsum16(float v) {
#pragma unroll
  for (int m = 1; m < 16; m <<= 1) v += __shfl_xor(v, m);
  return v;
}

// ---------------------------------------------------------------------------
// Kernel 1: downsample (::2,::2) + fp32 -> bf16, store token-major xsT[b][n][c]
// ---------------------------------------------------------------------------
__global__ void __launch_bounds__(256) pack_x_kernel(const float* __restrict__ x,
                                                     bf16_t* __restrict__ xsT) {
  int idx = blockIdx.x * 256 + threadIdx.x;          // over BATCH*NTOK*CIN
  int c = idx % CIN;
  int n = (idx / CIN) % NTOK;
  int b = idx / (CIN * NTOK);
  int y = n / HS, xq = n % HS;
  float v = x[(((size_t)b * CIN + c) * 96 + 2 * y) * 96 + 2 * xq];
  xsT[idx] = f2bf(v);
}

__global__ void __launch_bounds__(256) pack_w_kernel(const float* __restrict__ w,
                                                     bf16_t* __restrict__ wbf) {
  int idx = blockIdx.x * 256 + threadIdx.x;          // over OCH*CIN
  wbf[idx] = f2bf(w[idx]);
}

// ---------------------------------------------------------------------------
// Kernel 2: QKV projection GEMM  qkv[o][n] = sum_c w[o][c] * xsT[b][n][c]
// One 16x16 output tile per wave, K-loop in steps of 32 (bf16 WMMA).
// Epilogue: 8 consecutive channels per lane -> one packed v8bf store for q/k.
// ---------------------------------------------------------------------------
__global__ void __launch_bounds__(256) qkv_gemm_kernel(const bf16_t* __restrict__ wbf,
                                                       const bf16_t* __restrict__ xsT,
                                                       bf16_t* __restrict__ qT,
                                                       bf16_t* __restrict__ kT,
                                                       bf16_t* __restrict__ vM) {
  const int lane = threadIdx.x & 31;
  const int wv   = threadIdx.x >> 5;
  const int half = lane >> 4;
  const int l15  = lane & 15;
  const int base8 = half * 8;

  int tile = blockIdx.x * 8 + wv;
  int nt = tile % (NTOK / 16);
  int ot = (tile / (NTOK / 16)) % (OCH / 16);
  int b  = tile / ((NTOK / 16) * (OCH / 16));

  const bf16_t* wrow = wbf + (size_t)(ot * 16 + l15) * CIN;
  const bf16_t* xrow = xsT + ((size_t)b * NTOK + nt * 16 + l15) * CIN;

  v8f c = vzero8();
#pragma unroll 4
  for (int k0 = 0; k0 < CIN; k0 += 32) {
    v8bf a0 = *(const v8bf*)(wrow + k0 + half * 16);
    v8bf a1 = *(const v8bf*)(wrow + k0 + half * 16 + 8);
    v16bf A = cat8(a0, a1);
    v16bf B = *(const v16bf*)(xrow + k0 + half * 16);
    c = __builtin_amdgcn_wmma_f32_16x16x32_bf16(false, A, false, B,
                                                (short)0, c, false, false);
  }

  // D layout: lane, VGPR j -> o = ot*16 + 8*half + j (8 consecutive channels),
  // n = nt*16 + l15. The 8-run never crosses a q/k/v 64-boundary.
  int n = nt * 16 + l15;
  int obase = ot * 16 + 8 * half;
  int head = obase / 192;
  int r0 = obase - head * 192;
  size_t bh = (size_t)(b * HEADS + head);
  v8bf pk;
#pragma unroll
  for (int j = 0; j < 8; ++j) pk[j] = f2bf(c[j]);
  if (r0 < 64) {
    *(v8bf*)(qT + (bh * NTOK + n) * HDIM + r0) = pk;
  } else if (r0 < 128) {
    *(v8bf*)(kT + (bh * NTOK + n) * HDIM + (r0 - 64)) = pk;
  } else {
#pragma unroll
    for (int j = 0; j < 8; ++j)
      vM[(bh * HDIM + (r0 - 128 + j)) * NTOK + n] = pk[j];
  }
}

// ---------------------------------------------------------------------------
// Kernel 3: flash attention. Block = 8 waves sharing one (b, head); each wave
// owns one 16-query tile. K/V tiles are double-buffered in LDS and staged with
// GLOBAL_LOAD_ASYNC_TO_LDS_B128; stage i+1 is issued before computing stage i,
// drained with s_wait_asynccnt 0x2 (2 async loads per thread per stage,
// in-order completion), so the DMA overlaps the 8 WMMAs + online softmax.
// ---------------------------------------------------------------------------
__global__ void __launch_bounds__(256) attn_kernel(const bf16_t* __restrict__ qT,
                                                   const bf16_t* __restrict__ kT,
                                                   const bf16_t* __restrict__ vM,
                                                   float* __restrict__ out) {
  __shared__ __align__(32) bf16_t ktile[2][32][HDIM];  // [buf][m][d]     8 KB
  __shared__ __align__(32) bf16_t vtile[2][HDIM][32];  // [buf][d][m]     8 KB
  __shared__ __align__(32) bf16_t pbuf[8][16][32];     // per-wave P tile 8 KB

  const int tid  = threadIdx.x;
  const int lane = tid & 31;
  const int wv   = tid >> 5;
  const int half = lane >> 4;
  const int l15  = lane & 15;
  const int base8 = half * 8;

  int bid = blockIdx.x;
  int qt8 = bid % QT8;
  int h   = (bid / QT8) % HEADS;
  int b   = bid / (QT8 * HEADS);
  int qt  = qt8 * 8 + wv;                 // this wave's 16-query tile
  size_t bh = (size_t)(b * HEADS + h);

  // ---- per-thread staging addresses (16 bytes per thread per tile)
  unsigned klds0 = (unsigned)(uintptr_t)(&ktile[0][0][0]) + tid * 16;
  const bf16_t* kgbase = kT + bh * NTOK * HDIM;       // + kb*64, contiguous 4KB
  int vd  = tid >> 2;                                  // 0..63
  int vch = tid & 3;                                   // 16B chunk
  unsigned vlds0 = (unsigned)(uintptr_t)(&vtile[0][vd][vch * 8]);
  const bf16_t* vgbase = vM + (bh * HDIM + vd) * NTOK + vch * 8;  // + kb

#define ISSUE_STAGE(kbn, bsel)                                                  \
  {                                                                             \
    unsigned long long kga = (unsigned long long)(uintptr_t)(kgbase +           \
        (size_t)(kbn) * HDIM) + (unsigned)tid * 16u;                            \
    unsigned long long vga = (unsigned long long)(uintptr_t)(vgbase + (kbn));   \
    unsigned kl = klds0 + (unsigned)(bsel) * (unsigned)sizeof(bf16_t[32][HDIM]);\
    unsigned vl = vlds0 + (unsigned)(bsel) * (unsigned)sizeof(bf16_t[HDIM][32]);\
    asm volatile("global_load_async_to_lds_b128 %0, %1, off"                    \
                 :: "v"(kl), "v"(kga) : "memory");                              \
    asm volatile("global_load_async_to_lds_b128 %0, %1, off"                    \
                 :: "v"(vl), "v"(vga) : "memory");                              \
  }

  // ---- A(q) tiles 16x32: lane row n = qt*16 + l15; K = d.
  const bf16_t* qrow = qT + (bh * NTOK + qt * 16 + l15) * HDIM;
  v16bf Aq0 = cat8(*(const v8bf*)(qrow + 0  + half * 16), *(const v8bf*)(qrow + 8  + half * 16));
  v16bf Aq1 = cat8(*(const v8bf*)(qrow + 32 + half * 16), *(const v8bf*)(qrow + 40 + half * 16));

  v8f acc[4];
#pragma unroll
  for (int t = 0; t < 4; ++t) acc[t] = vzero8();
  float mrow[8], lrow[8];
#pragma unroll
  for (int j = 0; j < 8; ++j) { mrow[j] = -1e30f; lrow[j] = 0.0f; }

  ISSUE_STAGE(0, 0);                       // prologue prefetch

  int buf = 0;
  for (int kb = 0; kb < NTOK; kb += 32, buf ^= 1) {
    int kbn = kb + 32 < NTOK ? kb + 32 : kb;   // clamped re-issue on last iter
    ISSUE_STAGE(kbn, buf ^ 1);
    asm volatile("s_wait_asynccnt 0x2" ::: "memory");  // drain current stage
    __syncthreads();                       // tiles visible to all waves

    // ---- S tiles from LDS: S[n][m], m split into two 16-wide halves
    v8f S0 = vzero8(), S1 = vzero8();
    {
      const bf16_t* krow0 = &ktile[buf][l15][0];
      v16bf B0 = *(const v16bf*)(krow0 + half * 16);        // d 0..31
      v16bf B1 = *(const v16bf*)(krow0 + 32 + half * 16);   // d 32..63
      S0 = __builtin_amdgcn_wmma_f32_16x16x32_bf16(false, Aq0, false, B0,
                                                   (short)0, S0, false, false);
      S0 = __builtin_amdgcn_wmma_f32_16x16x32_bf16(false, Aq1, false, B1,
                                                   (short)0, S0, false, false);
    }
    {
      const bf16_t* krow1 = &ktile[buf][16 + l15][0];
      v16bf B0 = *(const v16bf*)(krow1 + half * 16);
      v16bf B1 = *(const v16bf*)(krow1 + 32 + half * 16);
      S1 = __builtin_amdgcn_wmma_f32_16x16x32_bf16(false, Aq0, false, B0,
                                                   (short)0, S1, false, false);
      S1 = __builtin_amdgcn_wmma_f32_16x16x32_bf16(false, Aq1, false, B1,
                                                   (short)0, S1, false, false);
    }

    // ---- online softmax; row n = j + 8*half, lane holds m = l15 / 16+l15
#pragma unroll
    for (int j = 0; j < 8; ++j) {
      float s0 = S0[j], s1 = S1[j];
      float mx = rmax16(fmaxf(s0, s1));
      float nm = fmaxf(mrow[j], mx);
      float sc = __expf(mrow[j] - nm);
      float p0 = __expf(s0 - nm);
      float p1 = __expf(s1 - nm);
      float ts = rsum16(p0 + p1);
      lrow[j] = lrow[j] * sc + ts;
      mrow[j] = nm;
#pragma unroll
      for (int t = 0; t < 4; ++t) acc[t][j] *= sc;
      pbuf[wv][j + 8 * half][l15]      = f2bf(p0);
      pbuf[wv][j + 8 * half][16 + l15] = f2bf(p1);
    }
    asm volatile("s_wait_dscnt 0x0" ::: "memory");   // our LDS stores -> our reads

    // ---- re-read P in A-operand layout (row n = l15, K = m)
    v16bf Ap = cat8(*(const v8bf*)(&pbuf[wv][l15][half * 16]),
                    *(const v8bf*)(&pbuf[wv][l15][half * 16 + 8]));

    // ---- O[n][d] += P[n][m] * v[d][m]  (B[K=m][N=d] from LDS vtile)
#pragma unroll
    for (int t = 0; t < 4; ++t) {
      const bf16_t* vrow = &vtile[buf][t * 16 + l15][half * 16];
      v16bf Bv = *(const v16bf*)vrow;
      acc[t] = __builtin_amdgcn_wmma_f32_16x16x32_bf16(false, Ap, false, Bv,
                                                       (short)0, acc[t], false, false);
    }
    __syncthreads();                       // done reading buf before re-issue
  }

  // ---- epilogue: normalize; 8 j-values per lane are consecutive tokens ->
  //      one v8f (2x b128) store per d-tile.
  float inv[8];
#pragma unroll
  for (int j = 0; j < 8; ++j) inv[j] = 1.0f / lrow[j];
#pragma unroll
  for (int t = 0; t < 4; ++t) {
    int d = t * 16 + l15;
    v8f o;
#pragma unroll
    for (int j = 0; j < 8; ++j) o[j] = acc[t][j] * inv[j];
    float* dst = out + ((size_t)b * 384 + h * HDIM + d) * NTOK + qt * 16 + 8 * half;
    *(v8f*)dst = o;
  }
#undef ISSUE_STAGE
}

// ---------------------------------------------------------------------------
extern "C" void kernel_launch(void* const* d_in, const int* in_sizes, int n_in,
                              void* d_out, int out_size, void* d_ws, size_t ws_size,
                              hipStream_t stream) {
  (void)in_sizes; (void)n_in; (void)out_size; (void)ws_size;
  const float* x = (const float*)d_in[0];
  const float* w = (const float*)d_in[1];
  float* out = (float*)d_out;

  auto align256 = [](size_t v) { return (v + 255) & ~(size_t)255; };
  char* ws = (char*)d_ws;
  size_t szXsT = (size_t)BATCH * NTOK * CIN * sizeof(bf16_t);          // 14.2 MB
  size_t szW   = (size_t)OCH * CIN * sizeof(bf16_t);                   // 0.9 MB
  size_t szQKV = (size_t)BATCH * HEADS * NTOK * HDIM * sizeof(bf16_t); // 14.2 MB each

  bf16_t* xsT = (bf16_t*)ws;                      ws += align256(szXsT);
  bf16_t* wbf = (bf16_t*)ws;                      ws += align256(szW);
  bf16_t* qT  = (bf16_t*)ws;                      ws += align256(szQKV);
  bf16_t* kT  = (bf16_t*)ws;                      ws += align256(szQKV);
  bf16_t* vM  = (bf16_t*)ws;

  pack_x_kernel<<<(BATCH * NTOK * CIN) / 256, 256, 0, stream>>>(x, xsT);
  pack_w_kernel<<<(OCH * CIN) / 256, 256, 0, stream>>>(w, wbf);
  qkv_gemm_kernel<<<(BATCH * (OCH / 16) * (NTOK / 16)) / 8, 256, 0, stream>>>(
      wbf, xsT, qT, kT, vM);
  attn_kernel<<<(BATCH * HEADS * (NTOK / 16)) / 8, 256, 0, stream>>>(
      qT, kT, vM, out);
}
